// Tiny_Att_46394236731817
// MI455X (gfx1250) — matrix operand order repaired and checked by reading, not compiled, not run on hardware
//
#include <hip/hip_runtime.h>
#include <hip/hip_bf16.h>

typedef __attribute__((ext_vector_type(16))) _Float16 v16h;
typedef __attribute__((ext_vector_type(8)))  float    v8f;

#define IN_F   768
#define D_ATT  64
#define TOK    64
#define LDA    72    // padded f16 row (halves); 144 B = 9*16 -> b128-aligned rows
#define LDSC   66    // padded f32 row (floats)

// Shared-memory carve (phase-aliased regions, all transitions barrier-separated)
#define OFF_WSC  9216            // after 64x72 f16 xs / yL tile
#define OFF_QL   36864           // region B start
#define OFF_KL   (36864 + 9216)
#define OFF_VL   (36864 + 18432)
#define SMEM_SZ  64512           // 36864 + 3*9216

union HalfPack { v16h v; unsigned u[8]; _Float16 h[16]; };

static __device__ __forceinline__ v8f wmma_f16(v16h a, v16h b, v8f c) {
  return __builtin_amdgcn_wmma_f32_16x16x32_f16(
      /*neg_a=*/false, a, /*neg_b=*/false, b,
      /*c_mod=*/(short)0, c, /*reuse_a=*/false, /*reuse_b=*/false);
}

// A fragment (16x32 f16): rows row0..row0+15, K = kbase..kbase+31.
// This code's contiguous k order: lanes 0-15 hold K 0..15; lanes 16-31 hold K 16..31.
static __device__ __forceinline__ v16h load_a16(_Float16 (*S)[LDA], int row0,
                                                int kbase, int lane) {
  int row  = row0 + (lane & 15);
  int kofs = (lane < 16) ? 0 : 16;
  HalfPack r;
#pragma unroll
  for (int j = 0; j < 4; ++j)
    r.u[j] = *(const unsigned*)&S[row][kbase + kofs + 2 * j];
#pragma unroll
  for (int j = 0; j < 4; ++j)
    r.u[4 + j] = *(const unsigned*)&S[row][kbase + 8 + kofs + 2 * j];
  return r.v;
}

// B fragment (32x16 f16) with B[k][n] = S[n0+n][k]  (row-per-output-column
// source: weight tiles stored [outFeat][k], or k^T).  Contiguous k order: lanes 0-15 hold
// K = kbase..+15, lanes 16-31 hold K = kbase+16..+31.
static __device__ __forceinline__ v16h load_bT(_Float16 (*S)[LDA], int n0,
                                               int kbase, int lane) {
  int col = n0 + (lane & 15);
  int kb  = kbase + ((lane < 16) ? 0 : 16);
  HalfPack r;
#pragma unroll
  for (int j = 0; j < 8; ++j)
    r.u[j] = *(const unsigned*)&S[col][kb + 2 * j];
  return r.v;
}

// B fragment with B[k][n] = S[k][n0+n]  (natural orientation, e.g. v)
static __device__ __forceinline__ v16h load_bN(_Float16 (*S)[LDA], int n0,
                                               int kbase, int lane) {
  int n  = n0 + (lane & 15);
  int kb = kbase + ((lane < 16) ? 0 : 16);
  HalfPack r;
#pragma unroll
  for (int j = 0; j < 16; ++j) r.h[j] = S[kb + j][n];
  return r.v;
}

// Cooperative f32 -> f16 stage into LDS: rows x 64 tile, float4 global loads.
static __device__ __forceinline__ void stage_f16(_Float16 (*dst)[LDA],
                                                 const float* __restrict__ src,
                                                 int srcLd, int rows, int tid) {
  int nquads = rows * 16;  // (rows*64)/4
  for (int i = tid; i < nquads; i += 256) {
    int r = i >> 4, c4 = (i & 15) * 4;
    float4 w = *(const float4*)(src + r * srcLd + c4);
    union { _Float16 h[2]; unsigned u; } p0, p1;
    p0.h[0] = (_Float16)w.x; p0.h[1] = (_Float16)w.y;
    p1.h[0] = (_Float16)w.z; p1.h[1] = (_Float16)w.w;
    unsigned* d = (unsigned*)&dst[r][c4];
    d[0] = p0.u;
    d[1] = p1.u;
  }
}

__global__ __launch_bounds__(256) void tiny_att_kernel(
    const float* __restrict__ x, const float* __restrict__ W1,
    const float* __restrict__ b1, const float* __restrict__ W2,
    const float* __restrict__ b2, float* __restrict__ out) {
  __shared__ __align__(16) unsigned char smem[SMEM_SZ];
  // phase 1:
  _Float16(*xs)[LDA]  = reinterpret_cast<_Float16(*)[LDA]>(smem);            // [64][72]
  _Float16(*wsc)[LDA] = reinterpret_cast<_Float16(*)[LDA]>(smem + OFF_WSC);  // [192][72]
  // phases 2-3 (aliases xs/wsc):
  float(*sc)[LDSC] = reinterpret_cast<float(*)[LDSC]>(smem);                 // [64][66]
  // phases 4-5 (aliases sc):
  _Float16(*yL)[LDA]  = reinterpret_cast<_Float16(*)[LDA]>(smem);            // [64][72]
  _Float16(*w2c)[LDA] = reinterpret_cast<_Float16(*)[LDA]>(smem + OFF_WSC);  // [192][72]
  // phases 1-4:
  _Float16(*qL)[LDA] = reinterpret_cast<_Float16(*)[LDA]>(smem + OFF_QL);
  _Float16(*kL)[LDA] = reinterpret_cast<_Float16(*)[LDA]>(smem + OFF_KL);
  _Float16(*vL)[LDA] = reinterpret_cast<_Float16(*)[LDA]>(smem + OFF_VL);

  const int tid  = threadIdx.x;
  const int lane = tid & 31;
  const int wv   = tid >> 5;        // wave 0..7
  const int win  = blockIdx.x;      // (b,s) window 0..2047
  const float* xw = x + (size_t)win * TOK * IN_F;
  float*       ow = out + (size_t)win * TOK * IN_F;

  const int m  = wv >> 1;           // M tile (16 tokens) 0..3
  const int nh = wv & 1;            // N half
  const v8f vzero = {0.f, 0.f, 0.f, 0.f, 0.f, 0.f, 0.f, 0.f};

  // ---------------- phase 1: qkv = x @ W1^T + b1 ----------------
  v8f acc[6];
#pragma unroll
  for (int t = 0; t < 6; ++t) acc[t] = vzero;

  for (int kc = 0; kc < 12; ++kc) {           // K chunks of 64
    __syncthreads();
    stage_f16(xs, xw + kc * 64, IN_F, TOK, tid);     // x chunk   64x64
    stage_f16(wsc, W1 + kc * 64, IN_F, 192, tid);    // W1 chunk 192x64
    __syncthreads();
#pragma unroll
    for (int kap = 0; kap < 2; ++kap) {
      v16h a = load_a16(xs, m * 16, kap * 32, lane);
#pragma unroll
      for (int t = 0; t < 6; ++t) {
        v16h b = load_bT(wsc, (nh * 6 + t) * 16, kap * 32, lane);
        acc[t] = wmma_f16(a, b, acc[t]);
      }
    }
  }
  // scatter q/k/v (+bias) to LDS as f16
#pragma unroll
  for (int t = 0; t < 6; ++t) {
    int f = (nh * 6 + t) * 16 + (lane & 15);
    float bias = b1[f];
    _Float16(*dst)[LDA] = (f < 64) ? qL : (f < 128) ? kL : vL;
    int fl = (f < 64) ? f : (f < 128) ? (f - 64) : (f - 128);
#pragma unroll
    for (int r = 0; r < 8; ++r) {
      int tok = m * 16 + r + ((lane >= 16) ? 8 : 0);
      dst[tok][fl] = (_Float16)(acc[t][r] + bias);
    }
  }
  __syncthreads();

  // ---------------- phase 2: logits = q k^T / 8 ----------------
#pragma unroll
  for (int tt = 0; tt < 2; ++tt) {
    int tile = wv * 2 + tt;         // 16 tiles, 2 per wave
    int tm = tile >> 2, tn = tile & 3;
    v8f c = vzero;
#pragma unroll
    for (int kap = 0; kap < 2; ++kap) {
      v16h a = load_a16(qL, tm * 16, kap * 32, lane);
      v16h b = load_bT(kL, tn * 16, kap * 32, lane);
      c = wmma_f16(a, b, c);
    }
#pragma unroll
    for (int r = 0; r < 8; ++r) {
      int tok = tm * 16 + r + ((lane >= 16) ? 8 : 0);
      sc[tok][tn * 16 + (lane & 15)] = c[r] * 0.125f;  // 1/sqrt(64)
    }
  }
  __syncthreads();

  // ---------------- phase 3: row softmax -> probs (f16) in qL ----------------
  if (tid < 64) {
    float mx = -3.0e38f;
    for (int j = 0; j < 64; ++j) mx = fmaxf(mx, sc[tid][j]);
    float s = 0.f;
    for (int j = 0; j < 64; ++j) {
      float e = __expf(sc[tid][j] - mx);
      sc[tid][j] = e;
      s += e;
    }
    float inv = 1.0f / s;
    for (int j = 0; j < 64; ++j) qL[tid][j] = (_Float16)(sc[tid][j] * inv);
  }
  __syncthreads();

  // ---------------- phase 4: y = probs @ v ----------------
#pragma unroll
  for (int tt = 0; tt < 2; ++tt) {
    int tile = wv * 2 + tt;
    int tm = tile >> 2, tn = tile & 3;
    v8f c = vzero;
#pragma unroll
    for (int kap = 0; kap < 2; ++kap) {
      v16h a = load_a16(qL, tm * 16, kap * 32, lane);
      v16h b = load_bN(vL, tn * 16, kap * 32, lane);
      c = wmma_f16(a, b, c);
    }
#pragma unroll
    for (int r = 0; r < 8; ++r) {
      int tok = tm * 16 + r + ((lane >= 16) ? 8 : 0);
      yL[tok][tn * 16 + (lane & 15)] = (_Float16)c[r];
    }
  }
  __syncthreads();

  // ---------------- phase 5: out = y @ W2^T + b2 ----------------
  v16h a0 = load_a16(yL, m * 16, 0, lane);
  v16h a1 = load_a16(yL, m * 16, 32, lane);
  for (int nc = 0; nc < 4; ++nc) {            // 4 chunks of 192 out-features
    __syncthreads();                          // protect w2c region reuse
    stage_f16(w2c, W2 + nc * 192 * D_ATT, D_ATT, 192, tid);
    __syncthreads();
#pragma unroll
    for (int t = 0; t < 6; ++t) {
      int ntl = nh * 6 + t;                   // local N tile in chunk
      v8f c = vzero;
      c = wmma_f16(a0, load_bT(w2c, ntl * 16, 0, lane), c);
      c = wmma_f16(a1, load_bT(w2c, ntl * 16, 32, lane), c);
      int o = nc * 192 + ntl * 16 + (lane & 15);
      float bias = b2[o];
#pragma unroll
      for (int r = 0; r < 8; ++r) {
        int tok = m * 16 + r + ((lane >= 16) ? 8 : 0);
        ow[tok * IN_F + o] = c[r] + bias;     // 16 lanes -> 16 consecutive floats
      }
    }
  }
}

extern "C" void kernel_launch(void* const* d_in, const int* in_sizes, int n_in,
                              void* d_out, int out_size, void* d_ws,
                              size_t ws_size, hipStream_t stream) {
  const float* x  = (const float*)d_in[0];
  const float* W1 = (const float*)d_in[1];
  const float* b1 = (const float*)d_in[2];
  const float* W2 = (const float*)d_in[3];
  const float* b2 = (const float*)d_in[4];
  float* out = (float*)d_out;
  (void)in_sizes; (void)n_in; (void)out_size; (void)d_ws; (void)ws_size;
  tiny_att_kernel<<<dim3(2048), dim3(256), 0, stream>>>(x, W1, b1, W2, b2, out);
}
